// XNOR_VGG_56693568307387
// MI455X (gfx1250) — 8-bit matrix operand order repaired and checked by reading, compiled, not run on hardware
//
#include <hip/hip_runtime.h>


typedef __attribute__((ext_vector_type(8))) int v8i;
typedef __attribute__((ext_vector_type(4))) int v4i;

union AFrag { v8i v; unsigned long long q[4]; };
union BFrag { v8i v; v4i h[2]; };

#define TM 128
#define TN 64
#define KB 64

__device__ __forceinline__ void asyncLoad16ToLds(unsigned ldsOff, unsigned long long gaddr) {
    asm volatile("global_load_async_to_lds_b128 %0, %1, off"
                 :: "v"(ldsOff), "v"(gaddr) : "memory");
}
__device__ __forceinline__ void waitAsync0() {
    asm volatile("s_wait_asynccnt 0x0" ::: "memory");
}

__global__ __launch_bounds__(256)
void binconv_wmma_kernel(const signed char* __restrict__ xs,
                         const signed char* __restrict__ wp,
                         const float* __restrict__ alpha,
                         const float* __restrict__ bias,
                         const float* __restrict__ gamma,
                         const float* __restrict__ beta,
                         const float* __restrict__ mean,
                         const float* __restrict__ var,
                         float* __restrict__ out,
                         int Cin, int Cout, int H, int W)
{
    __shared__ signed char sA[2][TM * KB];

    const int tid  = threadIdx.x;
    const int wave = tid >> 5;
    const int lane = tid & 31;
    const int col  = lane & 15;
    const int half = lane >> 4;

    const int HW      = H * W;
    const int Ntot    = 4 * HW;
    const int n0      = blockIdx.x * TN;
    const int m0      = blockIdx.y * TM;
    const int Ktot    = 9 * Cin;
    const int cblocks = Cin / KB;
    const int nKb     = 9 * cblocks;

    int pB[4], pY[4], pX[4], pOK[4];
    #pragma unroll
    for (int j = 0; j < 4; ++j) {
        int n = n0 + j * 16 + col;
        pOK[j] = (n < Ntot);
        int nn = pOK[j] ? n : 0;
        pB[j] = nn / HW; int r = nn - pB[j] * HW;
        pY[j] = r / W;   pX[j] = r - pY[j] * W;
    }

    auto stageA = [&](int kb, int buf) {
        #pragma unroll
        for (int i = 0; i < 2; ++i) {
            int chunk = tid + i * 256;
            int row   = chunk >> 2;
            int boff  = (chunk & 3) * 16;
            int gm    = m0 + row; if (gm >= Cout) gm = Cout - 1;
            unsigned long long g = (unsigned long long)(wp + (long long)gm * Ktot + (long long)kb * KB + boff);
            unsigned ldsOff = (unsigned)(unsigned long long)(const void*)&sA[buf][row * KB + boff];
            asyncLoad16ToLds(ldsOff, g);
        }
    };

    const signed char* bPtr[4];
    int bOK[4];
    auto updateTap = [&](int kpos) {
        int ky = kpos / 3 - 1;
        int kx = kpos - (kpos / 3) * 3 - 1;
        #pragma unroll
        for (int j = 0; j < 4; ++j) {
            int iy = pY[j] + ky, ix = pX[j] + kx;
            bOK[j]  = pOK[j] && iy >= 0 && iy < H && ix >= 0 && ix < W;
            bPtr[j] = xs + ((long long)(pB[j] * H + iy) * W + ix) * Cin + half * 16;
        }
    };

    v8i acc[4];
    const v8i vz = {0, 0, 0, 0, 0, 0, 0, 0};
    #pragma unroll
    for (int j = 0; j < 4; ++j) acc[j] = vz;

    stageA(0, 0);
    int kpos = 0, cblk = 0;
    updateTap(0);

    for (int kb = 0; kb < nKb; ++kb) {
        const int buf = kb & 1;
        waitAsync0();
        __syncthreads();
        if (kb + 1 < nKb) stageA(kb + 1, buf ^ 1);

        AFrag a;
        const signed char* ab = &sA[buf][(wave * 16 + col) * KB + half * 16];
        a.q[0] = *(const unsigned long long*)(ab +  0);
        a.q[1] = *(const unsigned long long*)(ab +  8);
        a.q[2] = *(const unsigned long long*)(ab + 32);
        a.q[3] = *(const unsigned long long*)(ab + 40);

        const int coff = cblk * KB;
        #pragma unroll
        for (int j = 0; j < 4; ++j) {
            BFrag b;
            if (bOK[j]) {
                const signed char* bp = bPtr[j] + coff;
                b.h[0] = *(const v4i*)(bp);
                b.h[1] = *(const v4i*)(bp + 32);
            } else {
                v4i z = {0, 0, 0, 0};
                b.h[0] = z; b.h[1] = z;
            }
            acc[j] = __builtin_amdgcn_wmma_i32_16x16x64_iu8(true, a.v, true, b.v, acc[j],
                                                            false, false);
        }

        if (++cblk == cblocks) { cblk = 0; if (++kpos < 9) updateTap(kpos); }
    }

    #pragma unroll
    for (int j = 0; j < 4; ++j) {
        int n = n0 + j * 16 + col;
        if (n < Ntot) {
            #pragma unroll
            for (int r = 0; r < 8; ++r) {
                int m = m0 + wave * 16 + r + half * 8;
                if (m < Cout) {
                    float y   = (float)acc[j][r] * alpha[m] + bias[m];
                    float inv = rsqrtf(var[m] + 1e-5f);
                    y = (y - mean[m]) * (inv * gamma[m]) + beta[m];
                    y = fmaxf(y, 0.0f);
                    out[((long long)(pB[j] * Cout + m) * H + pY[j]) * W + pX[j]] = y;
                }
            }
        }
    }
}

__global__ __launch_bounds__(256)
void fc_bin_wmma_kernel(const signed char* __restrict__ Bs,
                        const signed char* __restrict__ Ws,
                        const float* __restrict__ alpha,
                        const float* __restrict__ bias,
                        float* __restrict__ out,
                        int O, int K, int doRelu)
{
    const int tid  = threadIdx.x;
    const int wave = tid >> 5;
    const int lane = tid & 31;
    const int col  = lane & 15;
    const int half = lane >> 4;
    const int m0   = blockIdx.x * TM + wave * 16;

    v8i acc = {0, 0, 0, 0, 0, 0, 0, 0};
    const signed char* arow = Ws + (long long)(m0 + col) * K + half * 16;
    const signed char* brow = Bs + (long long)col * K + half * 16;

    for (int k = 0; k < K; k += 2 * KB) {
        #pragma unroll
        for (int u = 0; u < 2; ++u) {
            const int kk = k + u * KB;
            AFrag a;
            a.q[0] = *(const unsigned long long*)(arow + kk +  0);
            a.q[1] = *(const unsigned long long*)(arow + kk +  8);
            a.q[2] = *(const unsigned long long*)(arow + kk + 32);
            a.q[3] = *(const unsigned long long*)(arow + kk + 40);
            BFrag b;
            b.h[0] = *(const v4i*)(brow + kk);
            b.h[1] = *(const v4i*)(brow + kk + 32);
            acc = __builtin_amdgcn_wmma_i32_16x16x64_iu8(true, a.v, true, b.v, acc, false, false);
        }
    }

    if (col < 4) {
        #pragma unroll
        for (int r = 0; r < 8; ++r) {
            int m = m0 + r + half * 8;
            if (m < O) {
                float y = (float)acc[r] * alpha[m] + bias[m];
                if (doRelu) y = fmaxf(y, 0.0f);
                out[(long long)col * O + m] = y;
            }
        }
    }
}


__global__ void sign_nhwc_kernel(const float* __restrict__ in, signed char* __restrict__ out,
                                 int C, int H, int W)
{
    long long total = 4LL * C * H * W;
    long long i = (long long)blockIdx.x * 256 + threadIdx.x;
    if (i >= total) return;
    int x = (int)(i % W); long long t = i / W;
    int y = (int)(t % H); t /= H;
    int c = (int)(t % C); int b = (int)(t / C);
    out[(((long long)b * H + y) * W + x) * C + c] = (in[i] >= 0.0f) ? 1 : -1;
}

__global__ __launch_bounds__(256)
void pack_wsign_kernel(const float* __restrict__ w, signed char* __restrict__ wp,
                       float* __restrict__ alpha, int I)
{
    __shared__ float red[256];
    const int o = blockIdx.x;
    const int K = 9 * I;
    float s = 0.0f;
    for (int idx = threadIdx.x; idx < K; idx += 256) {
        int ci = idx / 9, kp = idx - ci * 9;
        float v = w[(long long)o * K + idx];
        s += fabsf(v);
        wp[(long long)o * K + kp * I + ci] = (v >= 0.0f) ? 1 : -1;
    }
    red[threadIdx.x] = s; __syncthreads();
    for (int off = 128; off > 0; off >>= 1) {
        if ((int)threadIdx.x < off) red[threadIdx.x] += red[threadIdx.x + off];
        __syncthreads();
    }
    if (threadIdx.x == 0) alpha[o] = red[0] / (float)K;
}

__global__ __launch_bounds__(256)
void pack_fc_wsign_kernel(const float* __restrict__ w, signed char* __restrict__ wp,
                          float* __restrict__ alpha, int K)
{
    __shared__ float red[256];
    const int o = blockIdx.x;
    float s = 0.0f;
    for (int idx = threadIdx.x; idx < K; idx += 256) {
        float v = w[(long long)o * K + idx];
        s += fabsf(v);
        wp[(long long)o * K + idx] = (v >= 0.0f) ? 1 : -1;
    }
    red[threadIdx.x] = s; __syncthreads();
    for (int off = 128; off > 0; off >>= 1) {
        if ((int)threadIdx.x < off) red[threadIdx.x] += red[threadIdx.x + off];
        __syncthreads();
    }
    if (threadIdx.x == 0) alpha[o] = red[0] / (float)K;
}

__global__ void pack_fcB_kernel(const float* __restrict__ x, signed char* __restrict__ Bs, int K)
{
    long long total = 16LL * K;
    long long i = (long long)blockIdx.x * 256 + threadIdx.x;
    if (i >= total) return;
    int row = (int)(i / K); long long k = i - (long long)row * K;
    Bs[i] = (row < 4) ? ((x[(long long)row * K + k] >= 0.0f) ? 1 : -1) : (signed char)0;
}

__global__ void maxpool_kernel(const float* __restrict__ in, float* __restrict__ out,
                               int C, int H, int W)
{
    int Ho = H >> 1, Wo = W >> 1;
    long long total = 4LL * C * Ho * Wo;
    long long i = (long long)blockIdx.x * 256 + threadIdx.x;
    if (i >= total) return;
    int xo = (int)(i % Wo); long long t = i / Wo;
    int yo = (int)(t % Ho); t /= Ho;
    int c  = (int)(t % C);  int b = (int)(t / C);
    const float* p = in + ((long long)(b * C + c) * H + yo * 2) * W + xo * 2;
    out[i] = fmaxf(fmaxf(p[0], p[1]), fmaxf(p[W], p[W + 1]));
}

__global__ void conv0_kernel(const float* __restrict__ x, const float* __restrict__ w,
                             const float* __restrict__ bias,
                             const float* __restrict__ gamma, const float* __restrict__ beta,
                             const float* __restrict__ mean,  const float* __restrict__ var,
                             float* __restrict__ out, int H, int W)
{
    long long total = 4LL * 64 * H * W;
    long long i = (long long)blockIdx.x * 256 + threadIdx.x;
    if (i >= total) return;
    int xp = (int)(i % W); long long t = i / W;
    int yp = (int)(t % H); t /= H;
    int o  = (int)(t % 64); int b = (int)(t / 64);
    float acc = bias[o];
    for (int c = 0; c < 3; ++c)
        #pragma unroll
        for (int ky = 0; ky < 3; ++ky) {
            int iy = yp + ky - 1; if (iy < 0 || iy >= H) continue;
            #pragma unroll
            for (int kx = 0; kx < 3; ++kx) {
                int ix = xp + kx - 1; if (ix < 0 || ix >= W) continue;
                acc += x[((long long)(b * 3 + c) * H + iy) * W + ix]
                     * w[((o * 3 + c) * 3 + ky) * 3 + kx];
            }
        }
    float inv = rsqrtf(var[o] + 1e-5f);
    acc = (acc - mean[o]) * (inv * gamma[o]) + beta[o];
    out[i] = fmaxf(acc, 0.0f);
}

__global__ void fc3_dense_kernel(const float* __restrict__ x, const float* __restrict__ w,
                                 const float* __restrict__ b, float* __restrict__ out,
                                 int O, int K)
{
    int i = blockIdx.x * 256 + threadIdx.x;
    if (i >= 4 * O) return;
    int n = i / O, j = i - n * O;
    float s = b[j];
    const float* xr = x + (long long)n * K;
    const float* wr = w + (long long)j * K;
    for (int k = 0; k < K; ++k) s += xr[k] * wr[k];
    out[i] = s;
}


extern "C" void kernel_launch(void* const* d_in, const int* in_sizes, int n_in,
                              void* d_out, int out_size, void* d_ws, size_t ws_size,
                              hipStream_t stream)
{
    (void)in_sizes; (void)n_in; (void)out_size; (void)ws_size;

    auto P = [&](int ci, int j) { return (const float*)d_in[1 + ci * 6 + j]; };
    const float* fc1_w = (const float*)d_in[79]; const float* fc1_b = (const float*)d_in[80];
    const float* fc2_w = (const float*)d_in[81]; const float* fc2_b = (const float*)d_in[82];
    const float* fc3_w = (const float*)d_in[83]; const float* fc3_b = (const float*)d_in[84];

    char* ws = (char*)d_ws; size_t off = 0;
    auto alloc = [&](size_t bytes) -> char* {
        off = (off + 255) & ~(size_t)255;
        char* p = ws + off; off += bytes; return p;
    };
    float*       actA     = (float*)alloc(4ull * 64 * 224 * 224 * sizeof(float));
    float*       actB     = (float*)alloc(4ull * 64 * 224 * 224 * sizeof(float));
    signed char* sAct     = (signed char*)alloc(4ull * 224 * 224 * 64);
    signed char* wPack    = (signed char*)alloc(512ull * 9 * 512);
    float*       alphaBuf = (float*)alloc(4096 * sizeof(float));
    signed char* fcW      = (signed char*)alloc(4096ull * 25088);
    signed char* fcBb     = (signed char*)alloc(16ull * 25088);
    float*       fcX1     = (float*)alloc(4ull * 4096 * sizeof(float));
    float*       fcX2     = (float*)alloc(4ull * 4096 * sizeof(float));

    static const int convCh[13][2] = {{3,64},{64,64},{64,128},{128,128},{128,256},{256,256},
                                      {256,256},{256,512},{512,512},{512,512},{512,512},
                                      {512,512},{512,512}};
    static const int seq[] = {1,-1, 2,3,-1, 4,5,6,-1, 7,8,9,-1, 10,11,12,-1};

    int H = 224, W = 224, C = 64;

    {
        long long total = 4LL * 64 * H * W;
        conv0_kernel<<<(unsigned)((total + 255) / 256), 256, 0, stream>>>(
            (const float*)d_in[0], P(0,0), P(0,1), P(0,2), P(0,3), P(0,4), P(0,5), actA, H, W);
    }
    float* cur = actA; float* nxt = actB;

    for (int e : seq) {
        if (e < 0) {
            long long total = 4LL * C * (H / 2) * (W / 2);
            maxpool_kernel<<<(unsigned)((total + 255) / 256), 256, 0, stream>>>(cur, nxt, C, H, W);
            float* t = cur; cur = nxt; nxt = t;
            H >>= 1; W >>= 1;
        } else {
            int Cin = convCh[e][0], Cout = convCh[e][1];
            long long totS = 4LL * Cin * H * W;
            sign_nhwc_kernel<<<(unsigned)((totS + 255) / 256), 256, 0, stream>>>(cur, sAct, Cin, H, W);
            pack_wsign_kernel<<<Cout, 256, 0, stream>>>(P(e,0), wPack, alphaBuf, Cin);
            dim3 g((unsigned)((4 * H * W + TN - 1) / TN), (unsigned)((Cout + TM - 1) / TM));
            binconv_wmma_kernel<<<g, 256, 0, stream>>>(
                sAct, wPack, alphaBuf, P(e,1), P(e,2), P(e,3), P(e,4), P(e,5),
                nxt, Cin, Cout, H, W);
            float* t = cur; cur = nxt; nxt = t;
            C = Cout;
        }
    }

    const int K1 = 512 * 7 * 7, O1 = 4096;
    pack_fcB_kernel<<<(unsigned)((16LL * K1 + 255) / 256), 256, 0, stream>>>(cur, fcBb, K1);
    pack_fc_wsign_kernel<<<O1, 256, 0, stream>>>(fc1_w, fcW, alphaBuf, K1);
    fc_bin_wmma_kernel<<<O1 / TM, 256, 0, stream>>>(fcBb, fcW, alphaBuf, fc1_b, fcX1, O1, K1, 1);

    const int K2 = 4096, O2 = 4096;
    pack_fcB_kernel<<<(unsigned)((16LL * K2 + 255) / 256), 256, 0, stream>>>(fcX1, fcBb, K2);
    pack_fc_wsign_kernel<<<O2, 256, 0, stream>>>(fc2_w, fcW, alphaBuf, K2);
    fc_bin_wmma_kernel<<<O2 / TM, 256, 0, stream>>>(fcBb, fcW, alphaBuf, fc2_b, fcX2, O2, K2, 1);

    fc3_dense_kernel<<<(4 * 1000 + 255) / 256, 256, 0, stream>>>(fcX2, fc3_w, fc3_b,
                                                                 (float*)d_out, 1000, 4096);
}
